// PaiConv_9629316677872
// MI455X (gfx1250) — matrix operand order repaired and checked by reading, not compiled, not run on hardware
//
#include <hip/hip_runtime.h>
#include <hip/hip_bf16.h>
#include <math.h>

typedef __attribute__((ext_vector_type(16))) _Float16 v16h;
typedef __attribute__((ext_vector_type(8)))  _Float16 v8h;
typedef __attribute__((ext_vector_type(8)))  float    v8f;

#define NN      32
#define CIN     64
#define TWOC    128
#define COUT    128
#define NPTS    2048
#define NBATCH  8
#define KDIM    4096          // TWOC * NN
#define TILE_M  16
#define AF_STRIDE 4104        // KDIM + 8 halves pad -> 8208 B row stride (16B-phase shifts per row)

// ---- dynamic LDS layout (bytes) ----
#define OFF_AF    0
#define SZ_AF     (TILE_M * AF_STRIDE * 2)   // 131328: A tile, f16 [16][4104]
#define OFF_SH    (OFF_AF + SZ_AF)
#define SZ_SH     (TWOC * NN * 4)            // 16384: shuffled feats, one point
#define OFF_P     (OFF_SH + SZ_SH)
#define SZ_P      (NN * 64 * 4)              // 8192: sin/cos RFF feats [32][64]
#define OFF_PERM  (OFF_P + SZ_P)
#define SZ_PERM   (NN * NN * 4)              // 4096
#define OFF_POS   (OFF_PERM + SZ_PERM)
#define SZ_POS    (NN * 8 * 4)               // 1024: pos7 rows (padded to 8)
#define OFF_REL   (OFF_POS + SZ_POS)
#define SZ_REL    (NN * 4 * 4)               // 512
#define OFF_NBC   (OFF_REL + SZ_REL)
#define SZ_NBC    (NN * 4 * 4)               // 512: neighbor coords
#define OFF_IDX   (OFF_NBC + SZ_NBC)
#define SZ_IDX    (NN * 4)                   // 128
#define SMEM_BYTES (OFF_IDX + SZ_IDX)        // 162176 B < 320KB/WGP

// Pre-convert conv_w [128][4096] fp32 -> f16 in workspace (L2-resident, shared by all tiles)
__global__ void convert_w_kernel(const float* __restrict__ w, _Float16* __restrict__ wf) {
    int i = blockIdx.x * blockDim.x + threadIdx.x;   // 524288 elements
    wf[i] = (_Float16)w[i];
}

__global__ __launch_bounds__(256) void paiconv_kernel(
    const float* __restrict__ x,        // [8,3,2048]
    const float* __restrict__ feat,     // [8,64,2048]
    const int*   __restrict__ nidx,     // [8,2048,32]
    const float* __restrict__ Brff,     // [7,32]
    const float* __restrict__ kern,     // [3,32]
    const float* __restrict__ mlp_w,    // [64,64]
    const float* __restrict__ mlp_b,    // [64]
    const _Float16* __restrict__ Wf,    // [128,4096] f16
    const float* __restrict__ conv_b,   // [128]
    float* __restrict__ out)            // [8,128,2048]
{
    extern __shared__ char smem[];
    _Float16* Af   = (_Float16*)(smem + OFF_AF);
    float*    sh   = (float*)(smem + OFF_SH);
    float*    P    = (float*)(smem + OFF_P);
    float*    perm = (float*)(smem + OFF_PERM);
    float*    pos7 = (float*)(smem + OFF_POS);
    float*    rel  = (float*)(smem + OFF_REL);
    float*    nbc  = (float*)(smem + OFF_NBC);
    int*      idxs = (int*)(smem + OFF_IDX);

    const int t    = threadIdx.x;
    const int tile = blockIdx.x;
    const float TWO_PI = 6.28318530717958647692f;

    // ---------------- Phase 1: build A tile (feats2 for 16 points, f16) ----------------
    for (int p = 0; p < TILE_M; ++p) {
        const int g = tile * TILE_M + p;
        const int b = g >> 11;                 // point g -> batch b, point n = g & 2047

        // step 0: gather neighbor indices + coords
        if (t < NN) {
            int id = nidx[(size_t)g * NN + t];
            idxs[t] = id;
            nbc[t*4+0] = x[((size_t)b*3 + 0) * NPTS + id];
            nbc[t*4+1] = x[((size_t)b*3 + 1) * NPTS + id];
            nbc[t*4+2] = x[((size_t)b*3 + 2) * NPTS + id];
        }
        __syncthreads();

        // step 1a: x_rel, x_dis, pos = 2*pi*[x_rep, x_rel, dis]   (x_rep = neighbor 0 coords)
        if (t < NN) {
            float rx = nbc[t*4+0] - nbc[0];
            float ry = nbc[t*4+1] - nbc[1];
            float rz = nbc[t*4+2] - nbc[2];
            float dis = sqrtf(rx*rx + ry*ry + rz*rz);
            rel[t*4+0] = rx; rel[t*4+1] = ry; rel[t*4+2] = rz;
            pos7[t*8+0] = TWO_PI * nbc[0];
            pos7[t*8+1] = TWO_PI * nbc[1];
            pos7[t*8+2] = TWO_PI * nbc[2];
            pos7[t*8+3] = TWO_PI * rx;
            pos7[t*8+4] = TWO_PI * ry;
            pos7[t*8+5] = TWO_PI * rz;
            pos7[t*8+6] = TWO_PI * dis;
        }
        __syncthreads();

        // step 1b: xf = pos @ B_rff ; P = [sin(xf) | cos(xf)]     (256 thr: i = t&31, 4 cols each)
        {
            int i  = t & 31;
            int j0 = (t >> 5) * 4;
            for (int jj = 0; jj < 4; ++jj) {
                int j = j0 + jj;
                float acc = 0.f;
                #pragma unroll
                for (int d = 0; d < 7; ++d) acc += pos7[i*8+d] * Brff[d*32 + j];
                P[i*64 + j]      = sinf(acc);
                P[i*64 + 32 + j] = cosf(acc);
            }
        }
        // step 2a: raw perm = relu(x_rel @ kernels + one_pad)     (independent of 1b outputs)
        {
            int i  = t & 31;
            int j0 = (t >> 5) * 4;
            for (int jj = 0; jj < 4; ++jj) {
                int j = j0 + jj;
                float v = rel[i*4+0]*kern[j] + rel[i*4+1]*kern[32+j] + rel[i*4+2]*kern[64+j];
                if (i == 0 && j == 0) v += 1.0f;
                perm[i*32 + j] = v > 0.f ? v : 0.f;
            }
        }
        __syncthreads();

        // step 2b: topkmax (normalize over rows i per column j, square, renormalize, threshold)
        if (t < NN) {
            int j = t;
            float s = 1e-6f;
            #pragma unroll
            for (int i = 0; i < NN; ++i) s += perm[i*32 + j];
            float r = 1.f / s;
            float s2 = 1e-6f;
            #pragma unroll
            for (int i = 0; i < NN; ++i) { float v = perm[i*32+j] * r; s2 += v*v; }
            float r2 = 1.f / s2;
            #pragma unroll
            for (int i = 0; i < NN; ++i) {
                float v = perm[i*32+j] * r; v = v*v*r2;
                perm[i*32+j] = (v > 0.1f) ? v : 0.f;
            }
        }
        // step 3 (concurrent with 2b; disjoint arrays): shuffled feats
        // sh[k][n] = feats[(k%4)*32 + k/4][n];  feats rows 0..63 = gathered features, 64..127 = MLP(P)
        {
            int nn_   = t & 31;
            int kbase = t >> 5;
            const int id0 = idxs[nn_];
            #pragma unroll
            for (int e = 0; e < 16; ++e) {
                int k = kbase + 8*e;
                int c = ((k & 3) << 5) + (k >> 2);     // inverse channel shuffle
                float val;
                if (c < CIN) {
                    val = feat[((size_t)b*CIN + c) * NPTS + id0];
                } else {
                    int o = c - CIN;
                    float a = mlp_b[o];
                    #pragma unroll 8
                    for (int cc = 0; cc < CIN; ++cc) a += mlp_w[o*CIN + cc] * P[nn_*64 + cc];
                    val = a;
                }
                sh[k*32 + nn_] = val;
            }
        }
        __syncthreads();

        // step 4: feats2 = sh @ perm -> A row for this point (f16), K index = c*32 + j
        {
            int j     = t & 31;
            int cbase = t >> 5;
            #pragma unroll
            for (int e = 0; e < 16; ++e) {
                int c = cbase + 8*e;
                float a = 0.f;
                #pragma unroll 8
                for (int i = 0; i < NN; ++i) a += sh[c*32 + i] * perm[i*32 + j];
                Af[p*AF_STRIDE + c*32 + j] = (_Float16)a;
            }
        }
        __syncthreads();
    }

    // ---------------- Phase 2: [16 x 4096] @ [4096 x 128] via v_wmma_f32_16x16x32_f16 ----------------
    {
        const int wave = t >> 5;           // 8 wave32 per block
        const int lane = t & 31;
        const int n0   = wave * 16;        // this wave's 16 output channels
        const int m    = lane & 15;
        const int hi   = lane >> 4;
        const int aoff = hi * 8;           // A: lanes<16 hold K 0-7/16-23; lanes>=16 hold K 8-15/24-31
        const int boff = hi * 8;           // B: lanes<16 hold K 0-7/16-23; lanes>=16 hold K 8-15/24-31
        v8f acc = {};
        const _Float16* arow = Af + m * AF_STRIDE;                 // LDS, 16B-aligned per lane
        const _Float16* brow = Wf + (size_t)(n0 + m) * KDIM;       // B column = conv_w row (B = W^T)
        for (int c = 0; c < TWOC; ++c) {
            const v8h* ap = (const v8h*)(arow + c*NN + aoff);
            v8h alo = ap[0];               // K = aoff + 0..7
            v8h ahi = ap[2];               // K = aoff + 16..23
            const v8h* bp = (const v8h*)(brow + c*NN + boff);
            v8h blo = bp[0];               // K = boff + 0..7
            v8h bhi = bp[2];               // K = boff + 16..23
            v16h a, bb;
            #pragma unroll
            for (int h = 0; h < 8; ++h) {
                a[h] = alo[h];  a[h+8] = ahi[h];
                bb[h] = blo[h]; bb[h+8] = bhi[h];
            }
            // D = A(16x32) * B(32x16) + C, f32 accumulate
            acc = __builtin_amdgcn_wmma_f32_16x16x32_f16(
                false, a, false, bb, (short)0, acc, false, false);
        }
        // C/D layout: VGPR r -> M = r + 8*(lane>=16), N = lane&15
        #pragma unroll
        for (int r = 0; r < 8; ++r) {
            int mm = r + hi*8;
            int ch = n0 + m;
            int g  = tile*TILE_M + mm;
            int b  = g >> 11;
            int n  = g & (NPTS - 1);
            out[((size_t)b*COUT + ch) * NPTS + n] = acc[r] + conv_b[ch];
        }
    }
}

extern "C" void kernel_launch(void* const* d_in, const int* in_sizes, int n_in,
                              void* d_out, int out_size, void* d_ws, size_t ws_size,
                              hipStream_t stream) {
    (void)in_sizes; (void)n_in; (void)out_size; (void)ws_size;
    const float* x      = (const float*)d_in[0];
    const float* feat   = (const float*)d_in[1];
    const int*   nidx   = (const int*)d_in[2];
    const float* Brff   = (const float*)d_in[3];
    const float* kern   = (const float*)d_in[4];
    const float* mlp_w  = (const float*)d_in[5];
    const float* mlp_b  = (const float*)d_in[6];
    const float* conv_w = (const float*)d_in[7];
    const float* conv_b = (const float*)d_in[8];
    float*    out = (float*)d_out;
    _Float16* Wf  = (_Float16*)d_ws;     // 1 MB f16 copy of conv_w

    convert_w_kernel<<<dim3((COUT*KDIM)/256), dim3(256), 0, stream>>>(conv_w, Wf);
    paiconv_kernel<<<dim3((NBATCH*NPTS)/TILE_M), dim3(256), SMEM_BYTES, stream>>>(
        x, feat, nidx, Brff, kern, mlp_w, mlp_b, Wf, conv_b, out);
}
